// Transformer_72473278152781
// MI455X (gfx1250) — compile-verified, matrix operand order checked by reading, not run on hardware
//
#include <hip/hip_runtime.h>

// ============================================================================
// MI455X (gfx1250, wave32) implementation.
// All heavy GEMMs use v_wmma_f32_16x16x32_bf16 (bf16 in, f32 accumulate).
// Activations & weights are converted to bf16 once in d_ws (~530 MB scratch).
// Per-block fusion: Wqb = Wb @ Wq so qb-projection folds into one GEMM.
// GEMM uses a register-staged software pipeline with double-buffered LDS:
//   iter k: issue global loads (k+1) -> regs | ds_load+WMMA on LDS[cur]
//           | store regs -> LDS[next];  one workgroup barrier per K-step.
// ============================================================================

typedef unsigned short u16;
typedef __attribute__((ext_vector_type(16))) __bf16 v16bf;
typedef __attribute__((ext_vector_type(8)))  float  v8f;

__device__ __forceinline__ float bf2f(u16 u) {
    return __uint_as_float(((unsigned)u) << 16);
}
__device__ __forceinline__ u16 f2bf(float f) {
    unsigned u = __float_as_uint(f);
    unsigned r = ((u >> 16) & 1u) + 0x7FFFu;   // round-to-nearest-even
    return (u16)((u + r) >> 16);
}

union Frag { uint4 u[2]; v16bf v; };

// ---------------------------------------------------------------------------
// f32 -> bf16 conversion, 4 elements/thread
// ---------------------------------------------------------------------------
__global__ void k_cvt(const float* __restrict__ src, u16* __restrict__ dst, int n4) {
    int i = blockIdx.x * blockDim.x + threadIdx.x;
    if (i < n4) {
        float4 v = reinterpret_cast<const float4*>(src)[i];
        u16* d = dst + (size_t)i * 4;
        d[0] = f2bf(v.x); d[1] = f2bf(v.y); d[2] = f2bf(v.z); d[3] = f2bf(v.w);
    }
}

// 512x512 transpose + convert (for Wq^T used in Wqb = Wb @ Wq)
__global__ void k_tr_cvt(const float* __restrict__ src, u16* __restrict__ dst) {
    int i = blockIdx.x * 256 + threadIdx.x;   // grid covers 512*512
    int r = i >> 9, c = i & 511;
    dst[c * 512 + r] = f2bf(src[i]);
}

// ---------------------------------------------------------------------------
// C[M,N] = A[M,K] * W[N,K]^T (+bias, +relu).  bf16 inputs, f32 accum.
// Tile: 128(M) x 64(N), 8 waves (256 threads), K-step 32.
// Double-buffered LDS + register-staged global loads (1 barrier / K-step).
// Requires M%128==0, N%64==0, K%32==0 (true for all uses here).
// ---------------------------------------------------------------------------
__global__ __launch_bounds__(256)
void k_gemm_nt(const u16* __restrict__ A, int lda,
               const u16* __restrict__ W, int ldw,
               const float* __restrict__ bias,
               u16* __restrict__ Cb, float* __restrict__ Cf, int ldc,
               int K, int relu)
{
    __shared__ u16 As[2][128][40];   // +8 pad keeps 16B alignment, spreads banks
    __shared__ u16 Bs[2][64][40];

    const int t    = threadIdx.x;
    const int m0   = blockIdx.y * 128;
    const int n0   = blockIdx.x * 64;
    const int wave = t >> 5, lane = t & 31;
    const int wm   = wave & 3, wn = wave >> 2;     // 4 M-waves x 2 N-waves
    const int lm   = lane & 15, lh = lane >> 4;

    // Per-thread staging assignment: A is 128x32 = 512 16B-chunks (2/thread),
    // B is 64x32 = 256 16B-chunks (1/thread).
    const int ra0 = t >> 2, ca = (t & 3) << 3;     // A chunk 0: rows 0..63
    const int ra1 = ra0 + 64;                      // A chunk 1: rows 64..127
    const u16* pa0 = A + (size_t)(m0 + ra0) * lda + ca;
    const u16* pa1 = A + (size_t)(m0 + ra1) * lda + ca;
    const u16* pw  = W + (size_t)(n0 + ra0) * ldw + ca;

    // ---- prologue: stage K-tile 0 (loads batched before stores) ----
    uint4 ta0 = *reinterpret_cast<const uint4*>(pa0);
    uint4 ta1 = *reinterpret_cast<const uint4*>(pa1);
    uint4 tb  = *reinterpret_cast<const uint4*>(pw);
    *reinterpret_cast<uint4*>(&As[0][ra0][ca]) = ta0;
    *reinterpret_cast<uint4*>(&As[0][ra1][ca]) = ta1;
    *reinterpret_cast<uint4*>(&Bs[0][ra0][ca]) = tb;

    v8f acc[2][2] = {};
    const int nk = K >> 5;

    for (int kk = 0; kk < nk; ++kk) {
        const int cur = kk & 1;
        const bool have_next = (kk + 1 < nk);
        uint4 na0 = {}, na1 = {}, nb = {};
        if (have_next) {                 // issue next tile's loads early
            const int ko = (kk + 1) << 5;
            na0 = *reinterpret_cast<const uint4*>(pa0 + ko);
            na1 = *reinterpret_cast<const uint4*>(pa1 + ko);
            nb  = *reinterpret_cast<const uint4*>(pw + ko);
            if (kk + 2 < nk)             // keep K+2 flowing into L2
                __builtin_prefetch(pa0 + ((kk + 2) << 5), 0, 1);
        }
        __syncthreads();                 // LDS[cur] ready; LDS[next] consumed

        // ---- fragments per ISA 16-bit A layout:
        // lane m=L&15, half=L>>4; v16bf = T[row][half*8..+8] ++ T[row][16+half*8..+8]
        Frag fa[2], fb[2];
#pragma unroll
        for (int mt = 0; mt < 2; ++mt) {
            int r = wm * 32 + mt * 16 + lm;
            fa[mt].u[0] = *reinterpret_cast<const uint4*>(&As[cur][r][lh * 8]);
            fa[mt].u[1] = *reinterpret_cast<const uint4*>(&As[cur][r][16 + lh * 8]);
        }
#pragma unroll
        for (int nt = 0; nt < 2; ++nt) {
            int r = wn * 32 + nt * 16 + lm;
            fb[nt].u[0] = *reinterpret_cast<const uint4*>(&Bs[cur][r][lh * 8]);
            fb[nt].u[1] = *reinterpret_cast<const uint4*>(&Bs[cur][r][16 + lh * 8]);
        }
#pragma unroll
        for (int mt = 0; mt < 2; ++mt)
#pragma unroll
            for (int nt = 0; nt < 2; ++nt)
                acc[mt][nt] = __builtin_amdgcn_wmma_f32_16x16x32_bf16(
                    false, fa[mt].v, false, fb[nt].v, (short)0, acc[mt][nt], false, false);

        if (have_next) {                 // s_wait_loadcnt lands here, after WMMA
            const int nxt = cur ^ 1;
            *reinterpret_cast<uint4*>(&As[nxt][ra0][ca]) = na0;
            *reinterpret_cast<uint4*>(&As[nxt][ra1][ca]) = na1;
            *reinterpret_cast<uint4*>(&Bs[nxt][ra0][ca]) = nb;
        }
    }

    // ---- epilogue: C/D layout: lane L -> N=L&15, M = (L>=16?8:0)+vgpr ----
#pragma unroll
    for (int mt = 0; mt < 2; ++mt) {
#pragma unroll
        for (int nt = 0; nt < 2; ++nt) {
            int n = n0 + wn * 32 + nt * 16 + lm;
            float bv = bias ? bias[n] : 0.f;
#pragma unroll
            for (int e = 0; e < 8; ++e) {
                int m = m0 + wm * 32 + mt * 16 + lh * 8 + e;
                float v = acc[mt][nt][e] + bv;
                if (relu) v = fmaxf(v, 0.f);
                if (Cf) Cf[(size_t)m * ldc + n] = v;
                if (Cb) Cb[(size_t)m * ldc + n] = f2bf(v);
            }
        }
    }
}

// ---------------------------------------------------------------------------
// ctx attention core (one workgroup per batch row b):
//   S[n][m] = (QB_b[n] . K_b[m]) * 512^-0.5       (WMMA, 64x64, K=512)
//   s[m]    = mean_n( mask[b,n,m] ? S : -1e9 )
//   a       = softmax(s);  x = a @ V_b  -> bf16 (B,512)
// ---------------------------------------------------------------------------
__global__ __launch_bounds__(128)
void k_ctx_attn(const u16* __restrict__ QB, const u16* __restrict__ Kp,
                const u16* __restrict__ Vp, const int* __restrict__ mask,
                u16* __restrict__ Xout)
{
    __shared__ float Sl[64][65];
    __shared__ float sm[64];

    const int b = blockIdx.x, t = threadIdx.x;
    const int w = t >> 5, lane = t & 31;
    const int lm = lane & 15, lh = lane >> 4;
    const u16* qb = QB + (size_t)b * 64 * 512;
    const u16* kb = Kp + (size_t)b * 64 * 512;
    const u16* vb = Vp + (size_t)b * 64 * 512;
    const float scale = 0.044194173824159216f;   // 512^-0.5

    v8f acc[4] = {};
    for (int k0 = 0; k0 < 512; k0 += 32) {
        Frag fa;
        {
            const u16* p = qb + (size_t)(w * 16 + lm) * 512 + k0 + lh * 8;
            fa.u[0] = *reinterpret_cast<const uint4*>(p);
            fa.u[1] = *reinterpret_cast<const uint4*>(p + 16);
        }
#pragma unroll
        for (int mt = 0; mt < 4; ++mt) {
            Frag fb;
            const u16* p = kb + (size_t)(mt * 16 + lm) * 512 + k0 + lh * 8;
            fb.u[0] = *reinterpret_cast<const uint4*>(p);
            fb.u[1] = *reinterpret_cast<const uint4*>(p + 16);
            acc[mt] = __builtin_amdgcn_wmma_f32_16x16x32_bf16(
                false, fa.v, false, fb.v, (short)0, acc[mt], false, false);
        }
    }
#pragma unroll
    for (int mt = 0; mt < 4; ++mt)
#pragma unroll
        for (int e = 0; e < 8; ++e)
            Sl[w * 16 + lh * 8 + e][mt * 16 + lm] = acc[mt][e] * scale;
    __syncthreads();

    if (t < 64) {                       // masked column mean (over query rows n)
        const int m = t;
        const int* mrow = mask + (size_t)b * 4096;
        float s = 0.f;
        for (int n = 0; n < 64; ++n)
            s += (mrow[n * 64 + m] == 0) ? -1e9f : Sl[n][m];
        sm[m] = s * (1.f / 64.f);
    }
    __syncthreads();
    if (t == 0) {                       // softmax over 64 (tiny; serial)
        float mx = -3.4e38f;
        for (int i = 0; i < 64; ++i) mx = fmaxf(mx, sm[i]);
        float ss = 0.f;
        for (int i = 0; i < 64; ++i) { float e = __expf(sm[i] - mx); Sl[0][i] = e; ss += e; }
        float inv = 1.f / ss;
        for (int i = 0; i < 64; ++i) sm[i] = Sl[0][i] * inv;
    }
    __syncthreads();

    float4 xa = {0.f, 0.f, 0.f, 0.f};   // x = a @ V, 4 cols per thread
    for (int m2 = 0; m2 < 64; ++m2) {
        float am = sm[m2];
        ushort4 vv = *reinterpret_cast<const ushort4*>(vb + (size_t)m2 * 512 + t * 4);
        xa.x += am * bf2f(vv.x); xa.y += am * bf2f(vv.y);
        xa.z += am * bf2f(vv.z); xa.w += am * bf2f(vv.w);
    }
    u16* xo = Xout + (size_t)b * 512 + t * 4;
    xo[0] = f2bf(xa.x); xo[1] = f2bf(xa.y); xo[2] = f2bf(xa.z); xo[3] = f2bf(xa.w);
}

// ---------------------------------------------------------------------------
// tri attention core (one 64-thread workgroup per b):
//   logits[n] = qb_b . K_b[n]   (no scale, matching reference)
//   a = softmax(logits);  x = a @ V_b -> bf16 (B,512)
// ---------------------------------------------------------------------------
__global__ __launch_bounds__(64)
void k_tri_attn(const u16* __restrict__ Qb, const u16* __restrict__ Kp,
                const u16* __restrict__ Vp, u16* __restrict__ Xout)
{
    __shared__ float ql[512];
    __shared__ float lg[64];
    const int b = blockIdx.x, t = threadIdx.x;

    const u16* q = Qb + (size_t)b * 512;
    for (int j = 0; j < 8; ++j) ql[t * 8 + j] = bf2f(q[t * 8 + j]);
    __syncthreads();

    const u16* kb = Kp + (size_t)b * 64 * 512;
    float s = 0.f;
    for (int k0 = 0; k0 < 512; k0 += 8) {
        ushort4 a = *reinterpret_cast<const ushort4*>(kb + (size_t)t * 512 + k0);
        ushort4 c = *reinterpret_cast<const ushort4*>(kb + (size_t)t * 512 + k0 + 4);
        s += ql[k0 + 0] * bf2f(a.x) + ql[k0 + 1] * bf2f(a.y)
           + ql[k0 + 2] * bf2f(a.z) + ql[k0 + 3] * bf2f(a.w)
           + ql[k0 + 4] * bf2f(c.x) + ql[k0 + 5] * bf2f(c.y)
           + ql[k0 + 6] * bf2f(c.z) + ql[k0 + 7] * bf2f(c.w);
    }
    lg[t] = s;
    __syncthreads();
    if (t == 0) {
        float mx = -3.4e38f;
        for (int i = 0; i < 64; ++i) mx = fmaxf(mx, lg[i]);
        float ss = 0.f;
        for (int i = 0; i < 64; ++i) { float e = __expf(lg[i] - mx); ql[i] = e; ss += e; }
        float inv = 1.f / ss;
        for (int i = 0; i < 64; ++i) lg[i] = ql[i] * inv;
    }
    __syncthreads();

    const u16* vbp = Vp + (size_t)b * 64 * 512;
    float xa[8] = {0, 0, 0, 0, 0, 0, 0, 0};
    for (int m = 0; m < 64; ++m) {
        float am = lg[m];
        const u16* vp = vbp + (size_t)m * 512 + t * 8;
        ushort4 a = *reinterpret_cast<const ushort4*>(vp);
        ushort4 c = *reinterpret_cast<const ushort4*>(vp + 4);
        xa[0] += am * bf2f(a.x); xa[1] += am * bf2f(a.y);
        xa[2] += am * bf2f(a.z); xa[3] += am * bf2f(a.w);
        xa[4] += am * bf2f(c.x); xa[5] += am * bf2f(c.y);
        xa[6] += am * bf2f(c.z); xa[7] += am * bf2f(c.w);
    }
    u16* xo = Xout + (size_t)b * 512 + t * 8;
    for (int j = 0; j < 8; ++j) xo[j] = f2bf(xa[j]);
}

// ---------------------------------------------------------------------------
// out = layernorm(ccl + concat(el,er)); split into L/R halves (f32 + opt bf16)
// ---------------------------------------------------------------------------
__global__ __launch_bounds__(256)
void k_ln(const float* __restrict__ ccl,
          const float* __restrict__ el, const float* __restrict__ er,
          const float* __restrict__ lnw, const float* __restrict__ lnb,
          float* __restrict__ outLf, float* __restrict__ outRf, int ldf,
          u16* __restrict__ outLb, u16* __restrict__ outRb)
{
    __shared__ float red[256];
    const int b = blockIdx.x, t = threadIdx.x;
    float v[4]; float s = 0.f;
#pragma unroll
    for (int j = 0; j < 4; ++j) {
        int i = t * 4 + j;
        float r = (i < 512) ? el[(size_t)b * 512 + i] : er[(size_t)b * 512 + i - 512];
        v[j] = ccl[(size_t)b * 1024 + i] + r;
        s += v[j];
    }
    red[t] = s; __syncthreads();
    for (int st = 128; st > 0; st >>= 1) { if (t < st) red[t] += red[t + st]; __syncthreads(); }
    float mu = red[0] * (1.f / 1024.f);
    __syncthreads();
    float sq = 0.f;
#pragma unroll
    for (int j = 0; j < 4; ++j) { float d = v[j] - mu; sq += d * d; }
    red[t] = sq; __syncthreads();
    for (int st = 128; st > 0; st >>= 1) { if (t < st) red[t] += red[t + st]; __syncthreads(); }
    float rstd = rsqrtf(red[0] * (1.f / 1024.f) + 1e-5f);
#pragma unroll
    for (int j = 0; j < 4; ++j) {
        int i = t * 4 + j;
        float y = (v[j] - mu) * rstd * lnw[i] + lnb[i];
        if (i < 512) {
            if (outLf) outLf[(size_t)b * ldf + i] = y;
            if (outLb) outLb[(size_t)b * 512 + i] = f2bf(y);
        } else {
            if (outRf) outRf[(size_t)b * ldf + i - 512] = y;
            if (outRb) outRb[(size_t)b * 512 + i - 512] = f2bf(y);
        }
    }
}

// ===========================================================================
// Host orchestration
// ===========================================================================
extern "C" void kernel_launch(void* const* d_in, const int* in_sizes, int n_in,
                              void* d_out, int out_size, void* d_ws, size_t ws_size,
                              hipStream_t stream)
{
    (void)in_sizes; (void)n_in; (void)out_size; (void)ws_size;
    const int B = 1024, C = 512, BN = 1024 * 64;

    const float* entL = (const float*)d_in[0];
    const float* entR = (const float*)d_in[1];
    const float* eel  = (const float*)d_in[2];
    const float* eer  = (const float*)d_in[3];
    const float* vhd  = (const float*)d_in[4];
    const float* vtl  = (const float*)d_in[5];
    const int*   maskL = (const int*)d_in[6];
    const int*   maskR = (const int*)d_in[7];

    // JAX pytree flatten: dicts sorted by key.  Per block (11 leaves):
    // 0:W1 1:W2 2:Wb 3:Wk 4:Wp/Wo 5:Wq 6:Wv 7:bp/bo 8:b1 9:ln_b 10:ln_w
    // "ctx" (4 blocks) precedes "ent" (4 blocks).
    auto P = [&](int grp, int blk, int leaf) -> const float* {
        return (const float*)d_in[8 + grp * 44 + blk * 11 + leaf];
    };

    char* wsp = (char*)d_ws; size_t off = 0;
    auto carve = [&](size_t bytes) -> void* {
        void* p = wsp + off; off = (off + bytes + 255) & ~(size_t)255; return p;
    };
    // bf16 activations (persistent): 4 x 64 MiB
    u16* eelb = (u16*)carve((size_t)BN * C * 2);
    u16* eerb = (u16*)carve((size_t)BN * C * 2);
    u16* vhb  = (u16*)carve((size_t)BN * C * 2);
    u16* vtb  = (u16*)carve((size_t)BN * C * 2);
    // projection outputs (reused): 3 x 64 MiB
    u16* p0 = (u16*)carve((size_t)BN * C * 2);
    u16* p1 = (u16*)carve((size_t)BN * C * 2);
    u16* p2 = (u16*)carve((size_t)BN * C * 2);
    // bf16 weights (per-block slots)
    u16* w1b  = (u16*)carve((size_t)1024 * 1024 * 2);
    u16* w2b  = (u16*)carve((size_t)1024 * 1024 * 2);
    u16* wbb  = (u16*)carve((size_t)512 * 512 * 2);
    u16* wkb  = (u16*)carve((size_t)512 * 512 * 2);
    u16* wpb  = (u16*)carve((size_t)512 * 512 * 2);
    u16* wvb  = (u16*)carve((size_t)512 * 512 * 2);
    u16* wqTb = (u16*)carve((size_t)512 * 512 * 2);
    u16* wqbb = (u16*)carve((size_t)512 * 512 * 2);
    // small transients
    u16*   xq   = (u16*)carve((size_t)B * C * 2);
    u16*   xat  = (u16*)carve((size_t)B * C * 2);
    u16*   catb = (u16*)carve((size_t)B * 1024 * 2);
    u16*   hb   = (u16*)carve((size_t)B * 1024 * 2);
    float* ccl  = (float*)carve((size_t)B * 1024 * 4);
    float* elS[2] = {(float*)carve((size_t)B * C * 4), (float*)carve((size_t)B * C * 4)};
    float* erS[2] = {(float*)carve((size_t)B * C * 4), (float*)carve((size_t)B * C * 4)};
    u16*   elb = (u16*)carve((size_t)B * C * 2);
    u16*   erb = (u16*)carve((size_t)B * C * 2);

    auto cvt = [&](const float* s, u16* d, int n) {
        k_cvt<<<(n / 4 + 255) / 256, 256, 0, stream>>>(s, d, n / 4);
    };
    auto gemm = [&](const u16* A, int lda, const u16* W, int ldw, const float* bias,
                    u16* Cb, float* Cf, int ldc, int M, int Nc, int K, int relu) {
        dim3 g(Nc / 64, M / 128);
        k_gemm_nt<<<g, 256, 0, stream>>>(A, lda, W, ldw, bias, Cb, Cf, ldc, K, relu);
    };

    // one-time bf16 conversion of the big activations
    cvt(eel, eelb, BN * C); cvt(eer, eerb, BN * C);
    cvt(vhd, vhb,  BN * C); cvt(vtl, vtb,  BN * C);

    auto prep_block = [&](int grp, int blk) {
        cvt(P(grp, blk, 0), w1b, 1024 * 1024);
        cvt(P(grp, blk, 1), w2b, 1024 * 1024);
        cvt(P(grp, blk, 2), wbb, 512 * 512);
        cvt(P(grp, blk, 3), wkb, 512 * 512);
        cvt(P(grp, blk, 4), wpb, 512 * 512);
        cvt(P(grp, blk, 6), wvb, 512 * 512);
        k_tr_cvt<<<1024, 256, 0, stream>>>(P(grp, blk, 5), wqTb);
        // Wqb = Wb @ Wq  =  Wb @ (Wq^T)^T   (512x512x512, WMMA)
        gemm(wbb, 512, wqTb, 512, nullptr, wqbb, nullptr, 512, 512, 512, 512, 0);
    };

    auto ctx_block = [&](int blk, const float* elf, const float* erf,
                         float* oLf, float* oRf, int ldf, u16* oLb, u16* oRb) {
        prep_block(0, blk);
        const float* bp = P(0, blk, 7);
        const float* b1 = P(0, blk, 8);
        // cl: q=eer, k=eel, v=V_head, mask_left
        gemm(eerb, 512, wqbb, 512, nullptr, p0, nullptr, 512, BN, 512, 512, 0);
        gemm(eelb, 512, wkb,  512, nullptr, p1, nullptr, 512, BN, 512, 512, 0);
        gemm(vhb,  512, wvb,  512, nullptr, p2, nullptr, 512, BN, 512, 512, 0);
        k_ctx_attn<<<B, 128, 0, stream>>>(p0, p1, p2, maskL, xat);
        gemm(xat, 512, wpb, 512, bp, catb, nullptr, 1024, B, 512, 512, 0);
        // cr: q=eel, k=eer, v=V_tail, mask_right
        gemm(eelb, 512, wqbb, 512, nullptr, p0, nullptr, 512, BN, 512, 512, 0);
        gemm(eerb, 512, wkb,  512, nullptr, p1, nullptr, 512, BN, 512, 512, 0);
        gemm(vtb,  512, wvb,  512, nullptr, p2, nullptr, 512, BN, 512, 512, 0);
        k_ctx_attn<<<B, 128, 0, stream>>>(p0, p1, p2, maskR, xat);
        gemm(xat, 512, wpb, 512, bp, catb + 512, nullptr, 1024, B, 512, 512, 0);
        // MLP + residual LN
        gemm(catb, 1024, w1b, 1024, b1, hb, nullptr, 1024, B, 1024, 1024, 1);
        gemm(hb,   1024, w2b, 1024, nullptr, nullptr, ccl, 1024, B, 1024, 1024, 0);
        k_ln<<<B, 256, 0, stream>>>(ccl, elf, erf, P(0, blk, 10), P(0, blk, 9),
                                    oLf, oRf, ldf, oLb, oRb);
    };

    auto ent_block = [&](int blk, const float* elf, const float* erf,
                         const u16* elbf, const u16* erbf,
                         float* oLf, float* oRf, int ldf, u16* oLb, u16* oRb) {
        prep_block(1, blk);
        const float* bo = P(1, blk, 7);
        const float* b1 = P(1, blk, 8);
        // hl: q=er, k=eel, v=V_head
        gemm(erbf, 512, wqbb, 512, nullptr, xq, nullptr, 512, B, 512, 512, 0);
        gemm(eelb, 512, wkb,  512, nullptr, p1, nullptr, 512, BN, 512, 512, 0);
        gemm(vhb,  512, wvb,  512, nullptr, p2, nullptr, 512, BN, 512, 512, 0);
        k_tri_attn<<<B, 64, 0, stream>>>(xq, p1, p2, xat);
        gemm(xat, 512, wpb, 512, bo, catb, nullptr, 1024, B, 512, 512, 0);
        // tl: q=el, k=eer, v=V_tail
        gemm(elbf, 512, wqbb, 512, nullptr, xq, nullptr, 512, B, 512, 512, 0);
        gemm(eerb, 512, wkb,  512, nullptr, p1, nullptr, 512, BN, 512, 512, 0);
        gemm(vtb,  512, wvb,  512, nullptr, p2, nullptr, 512, BN, 512, 512, 0);
        k_tri_attn<<<B, 64, 0, stream>>>(xq, p1, p2, xat);
        gemm(xat, 512, wpb, 512, bo, catb + 512, nullptr, 1024, B, 512, 512, 0);
        // MLP + residual LN
        gemm(catb, 1024, w1b, 1024, b1, hb, nullptr, 1024, B, 1024, 1024, 1);
        gemm(hb,   1024, w2b, 1024, nullptr, nullptr, ccl, 1024, B, 1024, 1024, 0);
        k_ln<<<B, 256, 0, stream>>>(ccl, elf, erf, P(1, blk, 10), P(1, blk, 9),
                                    oLf, oRf, ldf, oLb, oRb);
    };

    // ----- layers 0..2 -----
    int cur = 0;
    const float* elf = entL;
    const float* erf = entR;
    for (int i = 0; i < 3; ++i) {
        ctx_block(i, elf, erf, elS[cur], erS[cur], 512, elb, erb);
        elf = elS[cur]; erf = erS[cur]; cur ^= 1;
        ent_block(i, elf, erf, elb, erb, elS[cur], erS[cur], 512, elb, erb);
        elf = elS[cur]; erf = erS[cur]; cur ^= 1;
    }

    // ----- final heads: d_out = [el | er | context_CL | context_TL] -----
    float* doutf = (float*)d_out;
    const size_t BC = (size_t)B * C;
    ctx_block(3, elf, erf, doutf + 2 * BC, doutf + 2 * BC + 512, 1024, nullptr, nullptr);
    ent_block(3, elf, erf, elb, erb, doutf + 4 * BC, doutf + 4 * BC + 512, 1024,
              nullptr, nullptr);
    hipMemcpyAsync(doutf,      elf, BC * 4, hipMemcpyDeviceToDevice, stream);
    hipMemcpyAsync(doutf + BC, erf, BC * 4, hipMemcpyDeviceToDevice, stream);
}
